// QuanvolutionPlus_65481071408781
// MI455X (gfx1250) — matrix operand order repaired and checked by reading, not compiled, not run on hardware
//
#include <hip/hip_runtime.h>
#include <math.h>

typedef __attribute__((ext_vector_type(16))) _Float16 v16h;
typedef __attribute__((ext_vector_type(8)))  float    v8f;

#define KPAD 800            // 784 padded to 25 chunks of K=32
#define NCHUNK 25
#define BN_EPS 1e-5f

__launch_bounds__(32)
__global__ void quanv_fused_kernel(const float* __restrict__ x,
                                   const float* __restrict__ conv_w,
                                   const float* __restrict__ bn_gamma,
                                   const float* __restrict__ bn_beta,
                                   const float* __restrict__ bn_mean,
                                   const float* __restrict__ bn_var,
                                   const float* __restrict__ var_params,
                                   const float* __restrict__ lin_w,
                                   const float* __restrict__ lin_b,
                                   float* __restrict__ out)
{
    __shared__ _Float16 tileA[16][KPAD];   // fused activations, 16 images x 800 (f16)
    __shared__ _Float16 tileW[16][KPAD];   // lin_w padded: [n][k], n<10 valid
    __shared__ float    sc[16][16];        // logits scratch

    const int lane = threadIdx.x;          // 0..31 (wave32)
    const int b0   = blockIdx.x * 16;      // image tile base

    // ---- stage weight matrix W^T (as [N][K], f16, zero-padded) ----
    for (int idx = lane; idx < 16 * KPAD; idx += 32) {
        int n = idx / KPAD, k = idx % KPAD;
        float v = (n < 10 && k < 784) ? lin_w[n * 784 + k] : 0.0f;
        tileW[n][k] = (_Float16)v;
    }
    // zero the K-padding columns of the activation tile
    for (int idx = lane; idx < 16 * (KPAD - 784); idx += 32) {
        int r = idx / (KPAD - 784), k = 784 + idx % (KPAD - 784);
        tileA[r][k] = (_Float16)0.0f;
    }

    // ---- uniform params into registers (scalar loads) ----
    float cw[4][9], inv[4], shf[4];
#pragma unroll
    for (int c = 0; c < 4; ++c) {
        inv[c] = bn_gamma[c] * rsqrtf(bn_var[c] + BN_EPS);
        shf[c] = bn_beta[c] - bn_mean[c] * inv[c];
#pragma unroll
        for (int q = 0; q < 9; ++q) cw[c][q] = conv_w[c * 9 + q];
    }
    float cv[4], sv[4];
#pragma unroll
    for (int q = 0; q < 4; ++q) {
        float v = var_params[q];
        cv[q] = cosf(0.5f * v);
        sv[q] = sinf(0.5f * v);
    }

    __syncthreads();

    // ---- Pass 1: conv(s=2,p=1) + BN + ReLU, channel-major flatten ----
    for (int t = lane; t < 16 * 196; t += 32) {
        int r = t / 196, pj = t % 196;
        int i = pj / 14, j = pj % 14;
        const float* img = x + (size_t)(b0 + r) * 784;
        int y0 = 2 * i - 1, x0 = 2 * j - 1;
        float pix[3][3];
#pragma unroll
        for (int dy = 0; dy < 3; ++dy)
#pragma unroll
            for (int dx = 0; dx < 3; ++dx) {
                int yy = y0 + dy, xx = x0 + dx;
                pix[dy][dx] = (yy >= 0 && yy < 28 && xx >= 0 && xx < 28)
                                  ? img[yy * 28 + xx] : 0.0f;
            }
#pragma unroll
        for (int c = 0; c < 4; ++c) {
            float h = 0.0f;
#pragma unroll
            for (int dy = 0; dy < 3; ++dy)
#pragma unroll
                for (int dx = 0; dx < 3; ++dx)
                    h = fmaf(pix[dy][dx], cw[c][3 * dy + dx], h);
            h = fmaxf(fmaf(h, inv[c], shf[c]), 0.0f);
            tileA[r][c * 196 + pj] = (_Float16)h;
        }
    }
    __syncthreads();

    // ---- Pass 2: quantum features per 2x2 patch, patch-major flatten, += ----
    for (int t = lane; t < 16 * 196; t += 32) {
        int r = t / 196, pj = t % 196;
        int i = pj / 14, j = pj % 14;
        const float* img = x + (size_t)(b0 + r) * 784;
        float p00 = img[(2 * i) * 28 + 2 * j];
        float p01 = img[(2 * i) * 28 + 2 * j + 1];
        float p10 = img[(2 * i + 1) * 28 + 2 * j];
        float p11 = img[(2 * i + 1) * 28 + 2 * j + 1];
        float ha = 0.5f * (p00 + p10), hb = 0.5f * (p01 + p11);
        float ca, sa, cb, sb;
        sincosf(ha, &sa, &ca);
        sincosf(hb, &sb, &cb);
        // |psi> = w0 (x) w1 (real state, 2x2)
        float s00 = ca * cb, s01 = ca * sb, s10 = sa * cb, s11 = sa * sb;
        // CNOT(0->1): swap row-1 entries
        float tm = s10; s10 = s11; s11 = tm;
        // RY(v0) wire0
        float n00 = cv[0] * s00 - sv[0] * s10, n10 = sv[0] * s00 + cv[0] * s10;
        float n01 = cv[0] * s01 - sv[0] * s11, n11 = sv[0] * s01 + cv[0] * s11;
        // RY(v1) wire1
        s00 = cv[1] * n00 - sv[1] * n01;  s01 = sv[1] * n00 + cv[1] * n01;
        s10 = cv[1] * n10 - sv[1] * n11;  s11 = sv[1] * n10 + cv[1] * n11;
        // CNOT(1->0): swap column-1 entries
        tm = s01; s01 = s11; s11 = tm;
        // RY(v2) wire0
        n00 = cv[2] * s00 - sv[2] * s10;  n10 = sv[2] * s00 + cv[2] * s10;
        n01 = cv[2] * s01 - sv[2] * s11;  n11 = sv[2] * s01 + cv[2] * s11;
        // RY(v3) wire1
        s00 = cv[3] * n00 - sv[3] * n01;  s01 = sv[3] * n00 + cv[3] * n01;
        s10 = cv[3] * n10 - sv[3] * n11;  s11 = sv[3] * n10 + cv[3] * n11;
        float z0 = s00 * s00 + s01 * s01 - s10 * s10 - s11 * s11;
        float z1 = s00 * s00 + s10 * s10 - s01 * s01 - s11 * s11;
        float x0q = 2.0f * (s00 * s10 + s01 * s11);
        float x1q = 2.0f * (s00 * s01 + s10 * s11);
        int base = 4 * pj;
        tileA[r][base + 0] = (_Float16)((float)tileA[r][base + 0] + z0);
        tileA[r][base + 1] = (_Float16)((float)tileA[r][base + 1] + z1);
        tileA[r][base + 2] = (_Float16)((float)tileA[r][base + 2] + x0q);
        tileA[r][base + 3] = (_Float16)((float)tileA[r][base + 3] + x1q);
    }
    __syncthreads();

    // ---- Phase B: 16x800 @ 800x16 GEMM via v_wmma_f32_16x16x32_f16 ----
    const int h = lane >> 4;     // half-wave
    const int m = lane & 15;     // row (A) / col (B)
    v8f acc = {};
    for (int kc = 0; kc < NCHUNK; ++kc) {
        int kb = kc * 32;
        union { v16h v; unsigned u[8]; } afr, bfr;
#pragma unroll
        for (int p = 0; p < 8; ++p) {
            // ISA 16-bit A layout: K = 16*(p/4) + 8*h + 2*(p%4) (+ e%2)
            int koffA = ((p < 4) ? (2 * p) : (16 + 2 * (p - 4))) + 8 * h;
            // ISA 16-bit B layout: K = 16*(p/4) + 8*h + 2*(p%4) (+ e%2)
            int koffB = ((p < 4) ? (2 * p) : (16 + 2 * (p - 4))) + 8 * h;
            afr.u[p] = *(const unsigned*)&tileA[m][kb + koffA];
            bfr.u[p] = *(const unsigned*)&tileW[m][kb + koffB];
        }
        acc = __builtin_amdgcn_wmma_f32_16x16x32_f16(
            false, afr.v, false, bfr.v, (short)0, acc, false, false);
    }
    // scatter D (VGPR q: M = q + 8h, N = lane%16) to LDS
#pragma unroll
    for (int q = 0; q < 8; ++q) sc[q + 8 * h][m] = acc[q];
    __syncthreads();

    // ---- bias + log_softmax over the 10 valid columns ----
    if (lane < 16) {
        float z[10];
        float mx = -3.0e38f;
#pragma unroll
        for (int n = 0; n < 10; ++n) {
            z[n] = sc[lane][n] + lin_b[n];
            mx = fmaxf(mx, z[n]);
        }
        float s = 0.0f;
#pragma unroll
        for (int n = 0; n < 10; ++n) s += expf(z[n] - mx);
        float lse = mx + logf(s);
        float* o = out + (size_t)(b0 + lane) * 10;
#pragma unroll
        for (int n = 0; n < 10; ++n) o[n] = z[n] - lse;
    }
}

extern "C" void kernel_launch(void* const* d_in, const int* in_sizes, int n_in,
                              void* d_out, int out_size, void* d_ws, size_t ws_size,
                              hipStream_t stream) {
    const float* x        = (const float*)d_in[0];
    const float* conv_w   = (const float*)d_in[1];
    const float* bn_gamma = (const float*)d_in[2];
    const float* bn_beta  = (const float*)d_in[3];
    const float* bn_mean  = (const float*)d_in[4];
    const float* bn_var   = (const float*)d_in[5];
    const float* var_p    = (const float*)d_in[6];
    const float* lin_w    = (const float*)d_in[7];
    const float* lin_b    = (const float*)d_in[8];
    float* out = (float*)d_out;

    const int B = in_sizes[0] / 784;   // 16384
    dim3 grid(B / 16), block(32);
    quanv_fused_kernel<<<grid, block, 0, stream>>>(
        x, conv_w, bn_gamma, bn_beta, bn_mean, bn_var, var_p, lin_w, lin_b, out);
}
